// GraphBuilding_31602369364206
// MI455X (gfx1250) — matrix operand order repaired and checked by reading, compiled, not run on hardware
//
#include <hip/hip_runtime.h>
#include <hip/hip_bf16.h>

typedef __attribute__((ext_vector_type(16))) __bf16 v16bf;
typedef __attribute__((ext_vector_type(8)))  __bf16 v8bf;
typedef __attribute__((ext_vector_type(8)))  float  v8f;

#define BATCH   2
#define NPTS    8192      // 32*32*8 patches per batch
#define FDIM    512       // 1*8*8*8
#define KSTEPS  16        // 512 / 32
#define NTILE   64        // 8192 / 128 column tiles
#define BIGF    3.0e38f

// ---------------------------------------------------------------------------
// Kernel A: patch extraction. One wave (32 lanes) per patch.
// Writes fp32 feats (d_out), bf16 feats (ws), and per-patch squared norm (ws).
// ---------------------------------------------------------------------------
__global__ __launch_bounds__(256) void extract_kernel(
    const float* __restrict__ img, float* __restrict__ feats,
    __bf16* __restrict__ fb16, float* __restrict__ sq)
{
    int t    = threadIdx.x;
    int wave = t >> 5, lane = t & 31;
    int p    = blockIdx.x * 8 + wave;           // patch id in [0, B*N)
    int b    = p >> 13;                         // / 8192
    int n    = p & 8191;
    int hb   = n >> 8;                          // n / (Wb*Db)
    int wb   = (n >> 3) & 31;
    int db   = n & 7;

    const float* src = img + (size_t)b * (256u * 256u * 64u)
                           + (size_t)(hb * 8) * (256u * 64u)
                           + (size_t)(wb * 8) * 64u + db * 8;
    float*  fo = feats + (size_t)p * FDIM;
    __bf16* bo = fb16  + (size_t)p * FDIM;

    float ssq = 0.f;
#pragma unroll
    for (int si = 0; si < 2; ++si) {
        int s = lane + si * 32;                 // segment = h*8 + w, 0..63
        int h = s >> 3, w = s & 7;
        const float4* sp = reinterpret_cast<const float4*>(src + (size_t)h * (256u * 64u) + w * 64);
        float4 x0 = sp[0], x1 = sp[1];
        // fp32 out (feature offset f = s*8 + d)
        float4* op = reinterpret_cast<float4*>(fo + s * 8);
        op[0] = x0; op[1] = x1;
        // bf16 out
        v8bf bv;
        bv[0] = (__bf16)x0.x; bv[1] = (__bf16)x0.y; bv[2] = (__bf16)x0.z; bv[3] = (__bf16)x0.w;
        bv[4] = (__bf16)x1.x; bv[5] = (__bf16)x1.y; bv[6] = (__bf16)x1.z; bv[7] = (__bf16)x1.w;
        *reinterpret_cast<v8bf*>(bo + s * 8) = bv;
        ssq += x0.x*x0.x + x0.y*x0.y + x0.z*x0.z + x0.w*x0.w
             + x1.x*x1.x + x1.y*x1.y + x1.z*x1.z + x1.w*x1.w;
    }
    // wave32 reduction
#pragma unroll
    for (int m = 16; m >= 1; m >>= 1) ssq += __shfl_xor(ssq, m, 32);
    if (lane == 0) sq[p] = ssq;
}

// ---------------------------------------------------------------------------
// Kernel Z: zero the adjacency region (float4 stores).
// ---------------------------------------------------------------------------
__global__ __launch_bounds__(256) void zero_kernel(float4* __restrict__ p, long long n4)
{
    long long i = (long long)blockIdx.x * 256 + threadIdx.x;
    if (i < n4) p[i] = float4{0.f, 0.f, 0.f, 0.f};
}

// ---------------------------------------------------------------------------
// Kernel B: fused bf16-WMMA Gram tile -> squared distance -> top-8 -> scatter.
// Workgroup = 16 rows x full 8192-column sweep. 8 waves, wave owns a 16-col
// stripe per 128-col tile. A row-block fragment lives in 128 VGPRs for the
// entire sweep (loaded once).
// ---------------------------------------------------------------------------
__global__ __launch_bounds__(256, 1) void knn_kernel(
    const __bf16* __restrict__ fb16, const float* __restrict__ sq,
    float* __restrict__ adj)
{
    __shared__ float sqLds[NPTS];           // 32 KB: all row norms of this batch
    __shared__ float tile[16][132];         // d2 tile 16 x 128 (+pad)
    __shared__ float mergeV[256 * 8];
    __shared__ int   mergeI[256 * 8];

    int t    = threadIdx.x;
    int wave = t >> 5, lane = t & 31;
    int wg   = blockIdx.x;                  // 0..1023
    int b    = wg >> 9;                     // / 512
    int r0   = (wg & 511) * 16;             // row-block base

    const __bf16* fbase = fb16 + (size_t)b * NPTS * FDIM;
    const float*  sqb   = sq + b * NPTS;

    for (int i = t; i < NPTS; i += 256) sqLds[i] = sqb[i];
    __syncthreads();

    int half = lane >> 4;                   // 0 or 1
    int l16  = lane & 15;

    // --- A fragments: row = r0 + l16; per 32-K step lane holds
    //     K = base+{0..7,16..23} (half 0) or base+{8..15,24..31} (half 1)
    v16bf a[KSTEPS];
    const __bf16* arow = fbase + (size_t)(r0 + l16) * FDIM;
#pragma unroll
    for (int s = 0; s < KSTEPS; ++s) {
        int kb = s * 32;
        v8bf lo = *reinterpret_cast<const v8bf*>(arow + kb + half * 8);
        v8bf hi = *reinterpret_cast<const v8bf*>(arow + kb + 16 + half * 8);
        a[s] = __builtin_shufflevector(lo, hi, 0,1,2,3,4,5,6,7,8,9,10,11,12,13,14,15);
    }

    float sqr[8];
#pragma unroll
    for (int v = 0; v < 8; ++v) sqr[v] = sqLds[r0 + v + half * 8];

    // per-thread sorted top-8 (ascending)
    float tv[8]; int ti[8];
#pragma unroll
    for (int j = 0; j < 8; ++j) { tv[j] = BIGF; ti[j] = -1; }

    int scanRow = t >> 4;                   // 0..15
    int scanC0  = (t & 15) * 8;             // 8 cols per thread per tile

#pragma unroll 1
    for (int tl = 0; tl < NTILE; ++tl) {
        int ct = tl * 128;
        int wc = ct + wave * 16 + l16;      // this lane's column
        const __bf16* brow = fbase + (size_t)wc * FDIM;

        v8f c = {};
#pragma unroll
        for (int s = 0; s < KSTEPS; ++s) {
            // B fragment: col = wc; K = base+{0..7,16..23} (half 0) or base+{8..15,24..31} (half 1)
            v8bf blo = *reinterpret_cast<const v8bf*>(brow + s * 32 + half * 8);
            v8bf bhi = *reinterpret_cast<const v8bf*>(brow + s * 32 + 16 + half * 8);
            v16bf bv = __builtin_shufflevector(blo, bhi, 0,1,2,3,4,5,6,7,8,9,10,11,12,13,14,15);
            c = __builtin_amdgcn_wmma_f32_16x16x32_bf16(
                    false, a[s], false, bv, (short)0, c, false, false);
        }

        float sqc = sqLds[wc];
#pragma unroll
        for (int v = 0; v < 8; ++v) {
            int m   = v + half * 8;
            float d2 = sqr[v] + sqc - 2.0f * c[v];
            if (r0 + m == wc) d2 = BIGF;    // exclude self
            tile[m][wave * 16 + l16] = d2;
        }
        __syncthreads();

        // 256 threads: thread scans 8 candidates of its row this tile
#pragma unroll
        for (int d = 0; d < 8; ++d) {
            float val = tile[scanRow][scanC0 + d];
            int   idx = ct + scanC0 + d;
            if (val < tv[7]) {
                tv[7] = val; ti[7] = idx;
#pragma unroll
                for (int j = 7; j > 0; --j) {
                    if (tv[j] < tv[j-1]) {
                        float fv = tv[j]; tv[j] = tv[j-1]; tv[j-1] = fv;
                        int   iv = ti[j]; ti[j] = ti[j-1]; ti[j-1] = iv;
                    }
                }
            }
        }
        __syncthreads();
    }

    // dump lists: row r's 128 candidates land in mergeV[r*128 .. r*128+127]
#pragma unroll
    for (int j = 0; j < 8; ++j) { mergeV[t * 8 + j] = tv[j]; mergeI[t * 8 + j] = ti[j]; }
    __syncthreads();

    if (t < 16) {
        int base = t * 128;
        int gr   = r0 + t;
        float* adjb = adj + (size_t)b * NPTS * NPTS;
        for (int j = 0; j < 8; ++j) {
            float best = BIGF; int bp = 0;
            for (int q = 0; q < 128; ++q) {
                float v = mergeV[base + q];
                if (v < best) { best = v; bp = q; }
            }
            int col = mergeI[base + bp];
            mergeV[base + bp] = BIGF;
            if (col >= 0) {
                // (A + A^T)/2: symmetric collisions sum to 1.0 via f32 atomics
                unsafeAtomicAdd(adjb + (size_t)gr  * NPTS + col, 0.5f);
                unsafeAtomicAdd(adjb + (size_t)col * NPTS + gr,  0.5f);
            }
        }
    }
}

// ---------------------------------------------------------------------------
extern "C" void kernel_launch(void* const* d_in, const int* in_sizes, int n_in,
                              void* d_out, int out_size, void* d_ws, size_t ws_size,
                              hipStream_t stream)
{
    const float* img = (const float*)d_in[0];

    float* feats = (float*)d_out;                                 // [2,8192,512]
    float* adj   = (float*)d_out + (size_t)BATCH * NPTS * FDIM;   // [2,8192,8192]

    __bf16* fb16 = (__bf16*)d_ws;                                 // 16 MB
    float*  sq   = (float*)((char*)d_ws + (size_t)BATCH * NPTS * FDIM * sizeof(__bf16));

    // 1) patch extraction + norms
    extract_kernel<<<(BATCH * NPTS) / 8, 256, 0, stream>>>(img, feats, fb16, sq);

    // 2) zero adjacency (134,217,728 floats = 33,554,432 float4)
    long long n4 = (long long)BATCH * NPTS * NPTS / 4;
    zero_kernel<<<(unsigned)((n4 + 255) / 256), 256, 0, stream>>>(
        reinterpret_cast<float4*>(adj), n4);

    // 3) fused WMMA gram / distance / top-8 / scatter
    knn_kernel<<<BATCH * (NPTS / 16), 256, 0, stream>>>(fb16, sq, adj);
}
